// TemporalGraphModelGCNBatch_66477503807889
// MI455X (gfx1250) — matrix operand order repaired and checked by reading, not compiled, not run on hardware
//
#include <hip/hip_runtime.h>
#include <hip/hip_bf16.h>

// ---------------------------------------------------------------------------
// TemporalGraphModelGCNBatch for MI455X (gfx1250, wave32, WMMA)
//   inputs (setup_inputs order):
//    0 x_edge   [400000,4] f32        6 w_hh  [256,64] f32
//    1 edge_idx [2,400000] i32        7 b_ih  [256]    f32
//    2 emb      [50000,128] f32       8 b_hh  [256]    f32
//    3 gcn_w    [128,128]  f32        9 pred_w[1,320]  f32
//    4 gcn_b    [128]      f32       10 pred_b[1]      f32
//    5 w_ih     [256,1]    f32       out: logits [400000] f32
// ---------------------------------------------------------------------------

#define N_NODES 50000
#define N_EDGES 400000
#define DIM     128
#define MTILES  3125          // 50000 / 16

typedef __attribute__((ext_vector_type(16))) __bf16 v16bf;
typedef __attribute__((ext_vector_type(8)))  float  v8f;

// ====================== Kernel A: xw = emb @ W^T (bf16 WMMA) ================
// A-frag (16x32 bf16): lanes 0-15 -> M=lane, K slots {0..7,16..23};
//                      lanes16-31 -> M=lane-16, K slots {8..15,24..31}.
// B-frag (32x16 bf16): lane -> N = nt*16+lane%16, slots -> K as in A-frag.
__global__ __launch_bounds__(256)
void k_xw(const float* __restrict__ emb, const float* __restrict__ gw,
          float* __restrict__ xw)
{
    __shared__ v16bf sB[32 * 32];            // frag f = nt*4+kc, 32 lanes each (32KB)
    const int tid = threadIdx.x, lane = tid & 31, w = tid >> 5;

    __bf16* bp = (__bf16*)sB;
    for (int i = tid; i < 32 * 32 * 16; i += 256) {
        int f = i >> 9, l = (i >> 4) & 31, s = i & 15;
        int nt = f >> 2, kc = f & 3;
        int K = kc * 32 + ((l >= 16) ? 8 : 0) + ((s < 8) ? s : (s + 8));
        bp[i] = (__bf16)gw[(nt * 16 + (l & 15)) * DIM + K];  // B(K,N) = W[N][K]
    }
    __syncthreads();

    int mt = blockIdx.x * 8 + w;
    if (mt >= MTILES) return;

    const int r0base = (lane >= 16) ? 8 : 0;
    const int M = mt * 16 + (lane & 15);
    const float* rowp = emb + (long)M * DIM;

    v16bf A[4];
#pragma unroll
    for (int kc = 0; kc < 4; ++kc) {
        int r0 = kc * 32 + r0base;
#pragma unroll
        for (int s = 0; s < 8; ++s) A[kc][s]     = (__bf16)rowp[r0 + s];
#pragma unroll
        for (int s = 0; s < 8; ++s) A[kc][8 + s] = (__bf16)rowp[r0 + 16 + s];
    }

#pragma unroll
    for (int nt = 0; nt < 8; ++nt) {
        v8f acc = {};
#pragma unroll
        for (int kc = 0; kc < 4; ++kc)
            acc = __builtin_amdgcn_wmma_f32_16x16x32_bf16(
                false, A[kc], false, sB[(nt * 4 + kc) * 32 + lane],
                (short)0, acc, false, false);
        const int row0 = mt * 16 + r0base;
        const int col  = nt * 16 + (lane & 15);
#pragma unroll
        for (int r = 0; r < 8; ++r)
            xw[(long)(row0 + r) * DIM + col] = acc[r];
    }
}

// ====================== degree / normalization / scatter ====================
__global__ void k_deg_init(float* deg) {
    int i = blockIdx.x * 256 + threadIdx.x;
    if (i < N_NODES) deg[i] = 1.0f;                 // self loop
}
__global__ void k_deg_count(const int* __restrict__ ei, float* deg) {
    int e = blockIdx.x * 256 + threadIdx.x;
    if (e < N_EDGES) atomicAdd(&deg[ei[N_EDGES + e]], 1.0f);
}
__global__ void k_dinv(float* deg) {
    int i = blockIdx.x * 256 + threadIdx.x;
    if (i < N_NODES) deg[i] = rsqrtf(deg[i]);
}
__global__ void k_zero(float* p, long n) {
    long i = (long)blockIdx.x * 256 + threadIdx.x;
    if (i < n) p[i] = 0.0f;
}
__global__ void k_scatter(const int* __restrict__ ei, const float* __restrict__ xw,
                          const float* __restrict__ dinv, float* __restrict__ gout) {
    long idx = (long)blockIdx.x * 256 + threadIdx.x;
    if (idx >= (long)N_EDGES * DIM) return;
    int e = (int)(idx >> 7), c = (int)(idx & 127);
    int s = ei[e], d = ei[N_EDGES + e];
    atomicAdd(&gout[(long)d * DIM + c], xw[(long)s * DIM + c] * dinv[s] * dinv[d]);
}
__global__ void k_self_bias(const float* __restrict__ xw, const float* __restrict__ dinv,
                            const float* __restrict__ gb, float* __restrict__ gout) {
    long idx = (long)blockIdx.x * 256 + threadIdx.x;
    if (idx >= (long)N_NODES * DIM) return;
    int i = (int)(idx >> 7), c = (int)(idx & 127);
    gout[idx] += xw[idx] * dinv[i] * dinv[i] + gb[c];
}

// ====================== Kernel D: edge LSTM (bf16 WMMA) + pw3 dot ===========
// Per wave: 16 edges. gates^T[256x16] = w_hh[256x64] @ h^T[64x16].
// w_hh pre-packed in LDS as 32 A-fragments (mt 0..15 x kc 0..1).
// h kept per-wave in LDS, edge-major bf16 [16][64] -> lane's B-frag = 2x uint4.
// C-layout (VGPR r <-> M row, lane <-> N edge) => each lane owns i/f/g/o rows
// for its own edge: activations happen entirely in registers.
__global__ __launch_bounds__(256)
void k_lstm(const float* __restrict__ x_edge, const float* __restrict__ whh,
            const float* __restrict__ wih, const float* __restrict__ bih,
            const float* __restrict__ bhh, const float* __restrict__ pw,
            const float* __restrict__ pb, float* __restrict__ out)
{
    __shared__ v16bf sA[32 * 32];      // 32 w_hh fragments (32KB)
    __shared__ uint4 sHq[8 * 128];     // per-wave h: [16 edges][64 K] bf16 (16KB)
    __shared__ float sWih[256], sBs[256], sPw3[64];
    __shared__ float sPb;

    const int tid = threadIdx.x, lane = tid & 31, w = tid >> 5;

    // pack w_hh -> A fragments (bf16)
    __bf16* ap = (__bf16*)sA;
    for (int i = tid; i < 32 * 32 * 16; i += 256) {
        int f = i >> 9, l = (i >> 4) & 31, s = i & 15;
        int mt = f >> 1, kc = f & 1;
        int K = kc * 32 + ((l >= 16) ? 8 : 0) + ((s < 8) ? s : (s + 8));
        int M = mt * 16 + (l & 15);
        ap[i] = (__bf16)whh[M * 64 + K];
    }
    for (int i = tid; i < 256; i += 256) { sWih[i] = wih[i]; sBs[i] = bih[i] + bhh[i]; }
    if (tid < 64) sPw3[tid] = pw[256 + tid];
    if (tid == 0) sPb = pb[0];
    uint4 z = make_uint4(0u, 0u, 0u, 0u);
    for (int i = tid; i < 8 * 128; i += 256) sHq[i] = z;   // h0 = 0
    __syncthreads();

    const int half8  = (lane >= 16) ? 8 : 0;
    const int edge_l = lane & 15;
    const long e = (long)blockIdx.x * 128 + w * 16 + edge_l;
    uint4*  hq = &sHq[w * 128];
    __bf16* hb = (__bf16*)hq;

    float c[32], h[32];
#pragma unroll
    for (int i = 0; i < 32; ++i) { c[i] = 0.f; h[i] = 0.f; }

    for (int t = 0; t < 4; ++t) {
        // B fragments from h LDS [16 edges][64 K] (lane<->edge N, slots<->K), 2x b128 each
        union { uint4 q[2]; v16bf v; } b0, b1;
        b0.q[0] = hq[edge_l * 8 + 0 + (lane >> 4)];
        b0.q[1] = hq[edge_l * 8 + 2 + (lane >> 4)];
        b1.q[0] = hq[edge_l * 8 + 4 + (lane >> 4)];
        b1.q[1] = hq[edge_l * 8 + 6 + (lane >> 4)];
        const float x = x_edge[e * 4 + t];

#pragma unroll
        for (int m0 = 0; m0 < 4; ++m0) {
            v8f acc[4];
#pragma unroll
            for (int g = 0; g < 4; ++g) {          // i,f,g,o gate tiles
                int mt = m0 + 4 * g;
                v8f a = {};
                a = __builtin_amdgcn_wmma_f32_16x16x32_bf16(
                        false, sA[(mt * 2 + 0) * 32 + lane], false, b0.v,
                        (short)0, a, false, false);
                a = __builtin_amdgcn_wmma_f32_16x16x32_bf16(
                        false, sA[(mt * 2 + 1) * 32 + lane], false, b1.v,
                        (short)0, a, false, false);
                acc[g] = a;
            }
#pragma unroll
            for (int r = 0; r < 8; ++r) {
                int j  = m0 * 16 + half8 + r;      // hidden unit index
                float iv = acc[0][r] + x * sWih[j]       + sBs[j];
                float fv = acc[1][r] + x * sWih[64 + j]  + sBs[64 + j];
                float gv = acc[2][r] + x * sWih[128 + j] + sBs[128 + j];
                float ov = acc[3][r] + x * sWih[192 + j] + sBs[192 + j];
                float is = 1.f / (1.f + __expf(-iv));
                float fs = 1.f / (1.f + __expf(-fv));
                float os = 1.f / (1.f + __expf(-ov));
                float gt = tanhf(gv);
                int ci = m0 * 8 + r;
                float cn = fs * c[ci] + is * gt;
                c[ci] = cn;
                float hn = os * tanhf(cn);
                h[ci] = hn;
                hb[edge_l * 64 + j] = (__bf16)hn;  // feed next step's B
            }
        }
    }

    // partial dot pred_w[256:320] . h  (each lane owns half the units)
    float partial = 0.f;
#pragma unroll
    for (int m0 = 0; m0 < 4; ++m0)
#pragma unroll
        for (int r = 0; r < 8; ++r)
            partial += h[m0 * 8 + r] * sPw3[m0 * 16 + half8 + r];
    partial += __shfl_xor(partial, 16, 32);
    if (lane < 16) out[e] = partial + sPb;
}

// ====================== Kernel E: += pw . [gcn[src], gcn[dst]] ==============
__global__ __launch_bounds__(256)
void k_pred(const float* __restrict__ gout, const int* __restrict__ ei,
            const float* __restrict__ pw, float* __restrict__ out)
{
    const int lane = threadIdx.x & 31, w = threadIdx.x >> 5;
    const long e = (long)blockIdx.x * 8 + w;       // exactly 400000 waves
    int s = ei[e], d = ei[N_EDGES + e];
    const float4* sr = (const float4*)(gout + (long)s * DIM);
    const float4* dr = (const float4*)(gout + (long)d * DIM);
    const float4* p1 = (const float4*)pw;
    const float4* p2 = (const float4*)(pw + DIM);
    float4 a = sr[lane], b = p1[lane];
    float4 cd = dr[lane], d2 = p2[lane];
    float v = a.x * b.x + a.y * b.y + a.z * b.z + a.w * b.w
            + cd.x * d2.x + cd.y * d2.y + cd.z * d2.z + cd.w * d2.w;
    for (int off = 16; off; off >>= 1) v += __shfl_down(v, off, 32);
    if (lane == 0) out[e] += v;
}

// ============================ launcher ======================================
extern "C" void kernel_launch(void* const* d_in, const int* in_sizes, int n_in,
                              void* d_out, int out_size, void* d_ws, size_t ws_size,
                              hipStream_t stream) {
    const float* x_edge = (const float*)d_in[0];
    const int*   ei     = (const int*)d_in[1];
    const float* emb    = (const float*)d_in[2];
    const float* gw     = (const float*)d_in[3];
    const float* gb     = (const float*)d_in[4];
    const float* wih    = (const float*)d_in[5];
    const float* whh    = (const float*)d_in[6];
    const float* bih    = (const float*)d_in[7];
    const float* bhh    = (const float*)d_in[8];
    const float* pw     = (const float*)d_in[9];
    const float* pb     = (const float*)d_in[10];
    float* out = (float*)d_out;

    float* ws   = (float*)d_ws;
    float* xw   = ws;                                   // 6.4M f32
    float* gout = ws + (long)N_NODES * DIM;             // 6.4M f32
    float* deg  = ws + 2L * N_NODES * DIM;              // 50k  f32

    // GCN linear (WMMA bf16)
    k_xw<<<(MTILES + 7) / 8, 256, 0, stream>>>(emb, gw, xw);
    // degree + normalization
    k_deg_init<<<(N_NODES + 255) / 256, 256, 0, stream>>>(deg);
    k_deg_count<<<(N_EDGES + 255) / 256, 256, 0, stream>>>(ei, deg);
    k_dinv<<<(N_NODES + 255) / 256, 256, 0, stream>>>(deg);
    // scatter-accumulate (L2-resident atomics)
    long gn = (long)N_NODES * DIM;
    k_zero<<<(int)((gn + 255) / 256), 256, 0, stream>>>(gout, gn);
    long sn = (long)N_EDGES * DIM;
    k_scatter<<<(int)((sn + 255) / 256), 256, 0, stream>>>(ei, xw, deg, gout);
    k_self_bias<<<(int)((gn + 255) / 256), 256, 0, stream>>>(xw, deg, gb, gout);
    // edge LSTM (WMMA bf16) writes base logits (h-dot + bias)
    k_lstm<<<N_EDGES / 128, 256, 0, stream>>>(x_edge, whh, wih, bih, bhh, pw, pb, out);
    // gather dots against GCN embeddings
    k_pred<<<N_EDGES / 8, 256, 0, stream>>>(gout, ei, pw, out);
}
